// GraphConv_12077448036551
// MI455X (gfx1250) — matrix operand order repaired and checked by reading, not compiled, not run on hardware
//
#include <hip/hip_runtime.h>

#define L_DIM 2048
#define D_DIM 256
#define BSZ   16
#define LPAD  40                      // padded LDS row stride (elements), 80B: 16B-aligned, 20-bank stride

typedef __attribute__((ext_vector_type(16))) __bf16 v16bf;
typedef __attribute__((ext_vector_type(8)))  __bf16 v8bf;
typedef __attribute__((ext_vector_type(8)))  float  v8f;

// ---- CDNA5 async global->LDS staging (ASYNCcnt path), guarded so the build
// ---- can never regress if the toolchain lacks the builtin.
#if defined(__HIP_DEVICE_COMPILE__) && __has_builtin(__builtin_amdgcn_global_load_async_to_lds_b128)
#define HAVE_ASYNC_LDS 1
#else
#define HAVE_ASYNC_LDS 0
#endif

#if HAVE_ASYNC_LDS
#if __has_builtin(__builtin_amdgcn_s_wait_asynccnt)
#define WAIT_ASYNC(N) __builtin_amdgcn_s_wait_asynccnt(N)
#else
#define WAIT_ASYNC(N) asm volatile("s_wait_asynccnt %0" ::"i"(N) : "memory")
#endif
// Exact parameter type per the compiler diagnostic: 16-byte int vector pointer,
// AS1 (global) source / AS3 (LDS) destination. Pointers formed via inttoptr:
// AS1 flat-global is identity-mapped; AS3 value = low 32 bits of the flat
// shared address = LDS byte offset.
typedef int b128_t __attribute__((vector_size(16)));
#define ASYNC_CP16(gptr, lptr)                                                  \
    __builtin_amdgcn_global_load_async_to_lds_b128(                             \
        (__attribute__((address_space(1))) b128_t*)(uintptr_t)(gptr),           \
        (__attribute__((address_space(3))) b128_t*)(uintptr_t)(lptr), 0, 0)
#endif

// ---------------------------------------------------------------------------
// 1) degree + dinv:  deg[l] = 1 + sum_m graph[l, m]  (graph symmetric)
// ---------------------------------------------------------------------------
__global__ void deg_kernel(const float* __restrict__ g, float* __restrict__ dinv) {
    __shared__ float red[256];
    const int l = blockIdx.x;
    float s = 0.f;
    for (int m = threadIdx.x; m < L_DIM; m += 256)
        s += g[(size_t)l * L_DIM + m];
    red[threadIdx.x] = s;
    __syncthreads();
    for (int off = 128; off > 0; off >>= 1) {
        if (threadIdx.x < off) red[threadIdx.x] += red[threadIdx.x + off];
        __syncthreads();
    }
    if (threadIdx.x == 0) dinv[l] = rsqrtf(red[0] + 1.0f);  // self-loop => deg >= 1
}

// ---------------------------------------------------------------------------
// 2) Abf = bf16(graph + I)
// ---------------------------------------------------------------------------
__global__ void prep_adj(const float* __restrict__ g, __bf16* __restrict__ Abf) {
    const size_t idx = (size_t)blockIdx.x * 256 + threadIdx.x;
    const int l = (int)(idx / L_DIM), m = (int)(idx % L_DIM);
    float v = g[idx] + (l == m ? 1.0f : 0.0f);
    Abf[idx] = (__bf16)v;
}

// ---------------------------------------------------------------------------
// 3) Wt[o][k] = bf16(W[k][o])   (K-major for contiguous B-fragment loads)
// ---------------------------------------------------------------------------
__global__ void prep_wt(const float* __restrict__ W, __bf16* __restrict__ Wt) {
    const int idx = blockIdx.x * 256 + threadIdx.x;
    const int k = idx / D_DIM, o = idx % D_DIM;
    Wt[(size_t)o * D_DIM + k] = (__bf16)W[idx];
}

// ---------------------------------------------------------------------------
// 4) h = x @ W via WMMA bf16; write ht[b][o][m] = bf16(dinv[m] * h[b][m][o])
//    Block: 256 thr = 8 waves; tile 32 rows x 256 cols; wave = 16 x 64.
//    Register double-buffered over K.
// ---------------------------------------------------------------------------
__global__ void xw_kernel(const float* __restrict__ x, const __bf16* __restrict__ Wt,
                          const float* __restrict__ dinv, __bf16* __restrict__ ht) {
    const int wave = threadIdx.x >> 5, lane = threadIdx.x & 31;
    const int half = lane >> 4, lr = lane & 15;
    const int wm = wave & 1, wn = wave >> 1;
    const int r0 = blockIdx.x * 32 + wm * 16;   // global row over B*L
    const int c0 = wn * 64;

    const int g = r0 + lr;
    const float* xrow = x + (size_t)g * D_DIM;
    const __bf16* wcol[4];
#pragma unroll
    for (int t = 0; t < 4; ++t)
        wcol[t] = Wt + (size_t)(c0 + 16 * t + lr) * D_DIM + half * 16;

    auto loadA = [&](int k0) -> v16bf {
        const float4* pa = (const float4*)(xrow + k0 + half * 16);
        const float4* pb = (const float4*)(xrow + k0 + half * 16 + 8);
        float4 f0 = pa[0], f1 = pa[1], f2 = pb[0], f3 = pb[1];
        v16bf a;
        a[0]=(__bf16)f0.x; a[1]=(__bf16)f0.y; a[2]=(__bf16)f0.z; a[3]=(__bf16)f0.w;
        a[4]=(__bf16)f1.x; a[5]=(__bf16)f1.y; a[6]=(__bf16)f1.z; a[7]=(__bf16)f1.w;
        a[8]=(__bf16)f2.x; a[9]=(__bf16)f2.y; a[10]=(__bf16)f2.z; a[11]=(__bf16)f2.w;
        a[12]=(__bf16)f3.x; a[13]=(__bf16)f3.y; a[14]=(__bf16)f3.z; a[15]=(__bf16)f3.w;
        return a;
    };

    v8f acc[4] = {};
    v16bf a_c = loadA(0);
    v16bf b_c[4];
#pragma unroll
    for (int t = 0; t < 4; ++t) b_c[t] = *(const v16bf*)(wcol[t]);

    for (int k0 = 0; k0 < D_DIM - 32; k0 += 32) {
        v16bf a_n = loadA(k0 + 32);
        v16bf b_n[4];
#pragma unroll
        for (int t = 0; t < 4; ++t) b_n[t] = *(const v16bf*)(wcol[t] + k0 + 32);
#pragma unroll
        for (int t = 0; t < 4; ++t)
            acc[t] = __builtin_amdgcn_wmma_f32_16x16x32_bf16(
                false, a_c, false, b_c[t], (short)0, acc[t], false, false);
        a_c = a_n;
#pragma unroll
        for (int t = 0; t < 4; ++t) b_c[t] = b_n[t];
    }
#pragma unroll
    for (int t = 0; t < 4; ++t)
        acc[t] = __builtin_amdgcn_wmma_f32_16x16x32_bf16(
            false, a_c, false, b_c[t], (short)0, acc[t], false, false);

#pragma unroll
    for (int t = 0; t < 4; ++t) {
        const int c = c0 + 16 * t + lr;
#pragma unroll
        for (int i = 0; i < 8; ++i) {
            const int gr = r0 + i + 8 * half;       // global row
            const int b = gr >> 11, m = gr & (L_DIM - 1);
            float v = acc[t][i] * dinv[m];
            ht[((size_t)b * D_DIM + c) * L_DIM + m] = (__bf16)v;
        }
    }
}

// ---------------------------------------------------------------------------
// 5) out[b,l,o] = relu(dinv[l] * sum_m (A+I)[l,m] * ht[b,o,m] + bias[o]) + x[b,l,o]
//    Block tile 64 (l) x 256 (o); 8 waves, wave tile 16 x 128 (8 accumulators).
//    Double-buffered LDS, staged with GLOBAL_LOAD_ASYNC_TO_LDS_B128 when the
//    toolchain exposes it (ASYNCcnt pipeline), else register staging.
// ---------------------------------------------------------------------------
__global__ void __launch_bounds__(256)
agg_kernel(const __bf16* __restrict__ Abf, const __bf16* __restrict__ ht,
           const float* __restrict__ dinv, const float* __restrict__ x,
           const float* __restrict__ bias, float* __restrict__ out) {
    __shared__ __bf16 As[2][64 * LPAD];     // 64 rows (l) x 32 k, padded
    __shared__ __bf16 Bs[2][256 * LPAD];    // 256 cols (o) x 32 k, padded

    const int tid = threadIdx.x;
    const int wave = tid >> 5, lane = tid & 31;
    const int half = lane >> 4, lr = lane & 15;
    const int wm = wave & 3, wn = wave >> 2;         // 4 row-groups x 2 col-groups
    const int r0 = blockIdx.x * 64;                  // block row (l) base
    const int b  = blockIdx.y;

    // staging assignments
    const int ar = tid >> 2;                         // A: row 0..63
    const int ac = (tid & 3) * 8;                    // A: 8-elem chunk of k
    const __bf16* agp = Abf + (size_t)(r0 + ar) * L_DIM + ac;
    const __bf16* hgp = ht + ((size_t)b * D_DIM + tid) * L_DIM;  // B: col = tid

    const int NSTEP = L_DIM / 32;
    v8f acc[8] = {};
    const int arow = (wm * 16 + lr) * LPAD;

#if HAVE_ASYNC_LDS
    // ---- async DMA pipeline: issue 5 b128 copies per thread per tile -------
    {
        // tile 0 -> LDS[0], tile 1 -> LDS[1]
        ASYNC_CP16(agp,            &As[0][ar * LPAD + ac]);
        ASYNC_CP16(hgp + 0,        &Bs[0][tid * LPAD + 0]);
        ASYNC_CP16(hgp + 8,        &Bs[0][tid * LPAD + 8]);
        ASYNC_CP16(hgp + 16,       &Bs[0][tid * LPAD + 16]);
        ASYNC_CP16(hgp + 24,       &Bs[0][tid * LPAD + 24]);
        ASYNC_CP16(agp + 32,       &As[1][ar * LPAD + ac]);
        ASYNC_CP16(hgp + 32,       &Bs[1][tid * LPAD + 0]);
        ASYNC_CP16(hgp + 40,       &Bs[1][tid * LPAD + 8]);
        ASYNC_CP16(hgp + 48,       &Bs[1][tid * LPAD + 16]);
        ASYNC_CP16(hgp + 56,       &Bs[1][tid * LPAD + 24]);
        WAIT_ASYNC(5);             // async ops retire in order: tile 0 landed
        __syncthreads();
    }
    for (int s = 0; s < NSTEP; ++s) {
        const int buf = s & 1;
        // compute: 8 WMMAs from the live buffer
        v8bf a0 = *(const v8bf*)(&As[buf][arow + half * 16]);
        v8bf a1 = *(const v8bf*)(&As[buf][arow + half * 16 + 8]);
        v16bf a = __builtin_shufflevector(a0, a1, 0,1,2,3,4,5,6,7,8,9,10,11,12,13,14,15);
#pragma unroll
        for (int t = 0; t < 8; ++t) {
            const int c = wn * 128 + 16 * t + lr;
            v8bf p0 = *(const v8bf*)(&Bs[buf][c * LPAD + half * 16]);
            v8bf p1 = *(const v8bf*)(&Bs[buf][c * LPAD + half * 16 + 8]);
            v16bf bf = __builtin_shufflevector(p0, p1, 0,1,2,3,4,5,6,7,8,9,10,11,12,13,14,15);
            acc[t] = __builtin_amdgcn_wmma_f32_16x16x32_bf16(
                false, a, false, bf, (short)0, acc[t], false, false);
        }
        __syncthreads();                      // all reads of `buf` done
        if (s + 2 < NSTEP) {
            const int m0 = (s + 2) * 32;      // refill `buf` with tile s+2
            ASYNC_CP16(agp + m0,       &As[buf][ar * LPAD + ac]);
            ASYNC_CP16(hgp + m0 + 0,   &Bs[buf][tid * LPAD + 0]);
            ASYNC_CP16(hgp + m0 + 8,   &Bs[buf][tid * LPAD + 8]);
            ASYNC_CP16(hgp + m0 + 16,  &Bs[buf][tid * LPAD + 16]);
            ASYNC_CP16(hgp + m0 + 24,  &Bs[buf][tid * LPAD + 24]);
            WAIT_ASYNC(5);                    // tile s+1 (issued earlier) landed
        } else {
            WAIT_ASYNC(0);                    // drain tail
        }
        __syncthreads();                      // publish buf^1 to all waves
    }
#else
    // ---- fallback: register-staged double buffer ---------------------------
    v8bf ga, gb0, gb1, gb2, gb3;
    ga  = *(const v8bf*)(agp);
    gb0 = *(const v8bf*)(hgp + 0);
    gb1 = *(const v8bf*)(hgp + 8);
    gb2 = *(const v8bf*)(hgp + 16);
    gb3 = *(const v8bf*)(hgp + 24);
    *(v8bf*)(&As[0][ar * LPAD + ac])  = ga;
    *(v8bf*)(&Bs[0][tid * LPAD + 0])  = gb0;
    *(v8bf*)(&Bs[0][tid * LPAD + 8])  = gb1;
    *(v8bf*)(&Bs[0][tid * LPAD + 16]) = gb2;
    *(v8bf*)(&Bs[0][tid * LPAD + 24]) = gb3;
    ga  = *(const v8bf*)(agp + 32);
    gb0 = *(const v8bf*)(hgp + 32);
    gb1 = *(const v8bf*)(hgp + 40);
    gb2 = *(const v8bf*)(hgp + 48);
    gb3 = *(const v8bf*)(hgp + 56);
    __syncthreads();

    for (int s = 0; s < NSTEP; ++s) {
        const int buf = s & 1;
        if (s + 1 < NSTEP) {
            const int nb = buf ^ 1;
            *(v8bf*)(&As[nb][ar * LPAD + ac])  = ga;
            *(v8bf*)(&Bs[nb][tid * LPAD + 0])  = gb0;
            *(v8bf*)(&Bs[nb][tid * LPAD + 8])  = gb1;
            *(v8bf*)(&Bs[nb][tid * LPAD + 16]) = gb2;
            *(v8bf*)(&Bs[nb][tid * LPAD + 24]) = gb3;
            if (s + 2 < NSTEP) {
                const int m0 = (s + 2) * 32;
                ga  = *(const v8bf*)(agp + m0);
                gb0 = *(const v8bf*)(hgp + m0 + 0);
                gb1 = *(const v8bf*)(hgp + m0 + 8);
                gb2 = *(const v8bf*)(hgp + m0 + 16);
                gb3 = *(const v8bf*)(hgp + m0 + 24);
            }
        }
        v8bf a0 = *(const v8bf*)(&As[buf][arow + half * 16]);
        v8bf a1 = *(const v8bf*)(&As[buf][arow + half * 16 + 8]);
        v16bf a = __builtin_shufflevector(a0, a1, 0,1,2,3,4,5,6,7,8,9,10,11,12,13,14,15);
#pragma unroll
        for (int t = 0; t < 8; ++t) {
            const int c = wn * 128 + 16 * t + lr;
            v8bf p0 = *(const v8bf*)(&Bs[buf][c * LPAD + half * 16]);
            v8bf p1 = *(const v8bf*)(&Bs[buf][c * LPAD + half * 16 + 8]);
            v16bf bf = __builtin_shufflevector(p0, p1, 0,1,2,3,4,5,6,7,8,9,10,11,12,13,14,15);
            acc[t] = __builtin_amdgcn_wmma_f32_16x16x32_bf16(
                false, a, false, bf, (short)0, acc[t], false, false);
        }
        __syncthreads();
    }
#endif

    // epilogue: dinv[l] scale + bias + relu + residual, fp32 out
#pragma unroll
    for (int t = 0; t < 8; ++t) {
        const int c = wn * 128 + 16 * t + lr;
        const float bo = bias[c];
#pragma unroll
        for (int i = 0; i < 8; ++i) {
            const int lrow = r0 + wm * 16 + i + 8 * half;
            float v = acc[t][i] * dinv[lrow] + bo;
            v = fmaxf(v, 0.0f);
            const size_t oi = ((size_t)b * L_DIM + lrow) * D_DIM + c;
            out[oi] = v + x[oi];
        }
    }
}

// ---------------------------------------------------------------------------
extern "C" void kernel_launch(void* const* d_in, const int* in_sizes, int n_in,
                              void* d_out, int out_size, void* d_ws, size_t ws_size,
                              hipStream_t stream) {
    const float* x     = (const float*)d_in[0];
    const float* graph = (const float*)d_in[1];
    const float* W     = (const float*)d_in[2];
    const float* bias  = (const float*)d_in[3];
    float* out = (float*)d_out;

    char* ws = (char*)d_ws;
    float*  dinv = (float*)ws;                                   //   8 KB
    __bf16* Wt   = (__bf16*)(ws + 8192);                         // 128 KB
    __bf16* Abf  = (__bf16*)(ws + 8192 + 131072);                //   8 MB
    __bf16* ht   = (__bf16*)(ws + 8192 + 131072 +
                             (size_t)L_DIM * L_DIM * 2);         //  16 MB

    deg_kernel<<<L_DIM, 256, 0, stream>>>(graph, dinv);
    prep_adj<<<(L_DIM * L_DIM) / 256, 256, 0, stream>>>(graph, Abf);
    prep_wt<<<(D_DIM * D_DIM) / 256, 256, 0, stream>>>(W, Wt);
    xw_kernel<<<(BSZ * L_DIM) / 32, 256, 0, stream>>>(x, Wt, dinv, ht);
    dim3 agrid(L_DIM / 64, BSZ);
    agg_kernel<<<agrid, 256, 0, stream>>>(Abf, ht, dinv, x, bias, out);
}
